// GPTNetwork_33036888440954
// MI455X (gfx1250) — compile-verified, matrix operand order checked by reading, not run on hardware
//
#include <hip/hip_runtime.h>
#include <math.h>

// ---------------------------------------------------------------------------
// CDNA5 (gfx1250) GPT forward pass.  wave32, WMMA bf16 16x16x32 + TDM staging.
// ---------------------------------------------------------------------------

typedef __attribute__((ext_vector_type(16))) __bf16 v16bf;
typedef __attribute__((ext_vector_type(8)))  float  v8f;
typedef __attribute__((ext_vector_type(4)))  unsigned int u32x4;
typedef __attribute__((ext_vector_type(8)))  int    i32x8;
typedef __attribute__((ext_vector_type(4)))  int    i32x4;

union FragB16 { v16bf v; u32x4 q[2]; };

// Model constants
constexpr int CB   = 16;    // batch
constexpr int CT   = 511;   // state tokens
constexpr int CS   = 512;   // total sequence
constexpr int CD   = 1024;  // model dim
constexpr int CH   = 16;    // heads
constexpr int CDH  = 64;    // head dim
constexpr int CL   = 8;     // layers
constexpr int CDFF = 4096;
constexpr int CHID = 100;
constexpr int COUT = 30;
constexpr int COBS = 60;

// ---------------------------------------------------------------------------
// TDM: issue one tensor DMA moving a (rows x 32) bf16 tile (row stride = K
// elements) from global memory into LDS.  D# layout per CDNA5 ISA ch.8.
// ---------------------------------------------------------------------------
__device__ __forceinline__
void tdm_load_tile_bf16(const __bf16* gptr, unsigned lds_off, int K, int rows)
{
    const unsigned long long ga = (unsigned long long)(size_t)gptr;
    u32x4 g0;
    g0[0] = 1u;                                               // count=1, user mode
    g0[1] = lds_off;                                          // lds_addr
    g0[2] = (unsigned)(ga & 0xFFFFFFFFu);                     // global_addr[31:0]
    g0[3] = (unsigned)((ga >> 32) & 0x1FFFFFFu) | (2u << 30); // addr[56:32] | type=2

    i32x8 g1;
    g1[0] = (int)(1u << 16);                                  // data_size=2B, no mask
    g1[1] = (int)(((unsigned)K & 0xFFFFu) << 16);             // tensor_dim0[15:0]
    g1[2] = (int)((((unsigned)K >> 16) & 0xFFFFu) |
                  (((unsigned)rows & 0xFFFFu) << 16));        // td0 hi | tensor_dim1 lo
    g1[3] = (int)(32u << 16);                                 // td1 hi=0 | tile_dim0=32
    g1[4] = (int)((unsigned)rows & 0xFFFFu);                  // tile_dim1 | tile_dim2=0
    g1[5] = (int)K;                                           // tensor_dim0_stride lo
    g1[6] = 0;                                                // stride hi | dim1_stride
    g1[7] = 0;

    i32x4 z4; z4[0] = z4[1] = z4[2] = z4[3] = 0;
#if defined(__clang_major__) && (__clang_major__ >= 23)
    i32x8 z8;
    #pragma unroll
    for (int i = 0; i < 8; ++i) z8[i] = 0;
    __builtin_amdgcn_tensor_load_to_lds(g0, g1, z4, z4, z8, 0);
#else
    __builtin_amdgcn_tensor_load_to_lds(g0, g1, z4, z4, 0);
#endif
}

// ---------------------------------------------------------------------------
// Embedding: x[b,s,:] = in[b,s,:] @ Wtok + btok + pos[s,:]
// ---------------------------------------------------------------------------
__global__ __launch_bounds__(256)
void embed_kernel(const float* __restrict__ states, const float* __restrict__ goals,
                  const float* __restrict__ Wtok, const float* __restrict__ btok,
                  const float* __restrict__ pos, float* __restrict__ x)
{
    const int row = blockIdx.x;            // b*CS + s
    const int b = row / CS, s = row % CS;
    __shared__ float ins[COBS];
    const float* src = (s == 0) ? (goals + (size_t)b * COBS)
                                : (states + ((size_t)b * CT + (s - 1)) * COBS);
    if (threadIdx.x < COBS) ins[threadIdx.x] = src[threadIdx.x];
    __syncthreads();
    #pragma unroll
    for (int i = 0; i < 4; ++i) {
        const int d = threadIdx.x + i * 256;
        float acc = btok[d] + pos[(size_t)s * CD + d];
        for (int o = 0; o < COBS; ++o) acc = fmaf(ins[o], Wtok[(size_t)o * CD + d], acc);
        x[(size_t)row * CD + d] = acc;
    }
}

// ---------------------------------------------------------------------------
// LayerNorm (D=1024) -> bf16 activations
// ---------------------------------------------------------------------------
__global__ __launch_bounds__(256)
void ln_kernel(const float* __restrict__ x, const float* __restrict__ g,
               const float* __restrict__ bta, __bf16* __restrict__ out)
{
    const int row = blockIdx.x;
    const size_t base = (size_t)row * CD;
    const int t = threadIdx.x;
    __shared__ float red[256];

    float s = 0.f;
    for (int i = t; i < CD; i += 256) s += x[base + i];
    red[t] = s; __syncthreads();
    for (int st = 128; st > 0; st >>= 1) { if (t < st) red[t] += red[t + st]; __syncthreads(); }
    const float mean = red[0] * (1.f / CD);
    __syncthreads();

    float vs = 0.f;
    for (int i = t; i < CD; i += 256) { float d = x[base + i] - mean; vs += d * d; }
    red[t] = vs; __syncthreads();
    for (int st = 128; st > 0; st >>= 1) { if (t < st) red[t] += red[t + st]; __syncthreads(); }
    const float rs = rsqrtf(red[0] * (1.f / CD) + 1e-5f);

    for (int i = t; i < CD; i += 256)
        out[base + i] = (__bf16)((x[base + i] - mean) * rs * g[i] + bta[i]);
}

// ---------------------------------------------------------------------------
// WMMA GEMM:  C(MxN) = A(MxK bf16, row-major) * B(KxN fp32->bf16, row-major)
// Workgroup tile 128x64, BK=32, 8 waves (4x2), each wave 32x32 (4 WMMAs/step).
// A tile staged by the Tensor Data Mover (TDM); B staged + converted by VALU.
// EP: 0 = bias -> bf16 ; 1 = bias + exact GELU -> bf16 ; 2 = bias + resid -> f32
// ---------------------------------------------------------------------------
template <int EP>
__global__ __launch_bounds__(256)
void gemm_bf16(const __bf16* __restrict__ A, const float* __restrict__ Bw,
               const float* __restrict__ bias, const float* __restrict__ resid,
               float* __restrict__ Cf, __bf16* __restrict__ Cb,
               int M, int N, int K)
{
    (void)M;
    __shared__ __bf16 As[128 * 32];  // [m][k]
    __shared__ __bf16 Bs[64 * 32];   // transposed: [n][k]

    const int tN = blockIdx.x * 64;
    const int tM = blockIdx.y * 128;
    const int t = threadIdx.x;
    const int wave = t >> 5, lane = t & 31;
    const int wm = wave >> 1, wn = wave & 1;
    const int lr = lane & 15;
    const int khalf = (lane < 16) ? 0 : 8;
    const unsigned lds_As = (unsigned)(size_t)(void*)As;

    v8f acc[2][2];
    #pragma unroll
    for (int m = 0; m < 2; ++m)
        #pragma unroll
        for (int n = 0; n < 2; ++n)
            #pragma unroll
            for (int i = 0; i < 8; ++i) acc[m][n][i] = 0.f;

    for (int k0 = 0; k0 < K; k0 += 32) {
        // --- A tile (128x32 bf16) via Tensor Data Mover, issued by wave 0
        if (wave == 0)
            tdm_load_tile_bf16(&A[(size_t)tM * K + k0], lds_As, K, 128);

        // --- stage B tile (32x64 fp32 -> bf16, transposed into [n][k])
        #pragma unroll
        for (int i = 0; i < 8; ++i) {
            const int idx = t + i * 256;         // 0..2047
            const int kk = idx >> 6, n = idx & 63;
            Bs[n * 32 + kk] = (__bf16)Bw[(size_t)(k0 + kk) * N + tN + n];
            if (k0 + 32 < K)
                __builtin_prefetch(&Bw[(size_t)(k0 + 32 + kk) * N + tN + n], 0, 1);
        }
        if (wave == 0) __builtin_amdgcn_s_wait_tensorcnt(0);
        __syncthreads();

        // --- fragments (ISA 16-bit A 16x32 layout: runs of 8 at kbase, kbase+16)
        FragB16 fa[2];
        #pragma unroll
        for (int m = 0; m < 2; ++m) {
            const __bf16* ap = &As[(wm * 32 + m * 16 + lr) * 32 + khalf];
            fa[m].q[0] = *(const u32x4*)(ap);
            fa[m].q[1] = *(const u32x4*)(ap + 16);
        }
        FragB16 fb[2];
        #pragma unroll
        for (int n = 0; n < 2; ++n) {
            const __bf16* bp = &Bs[(wn * 32 + n * 16 + lr) * 32 + khalf];
            fb[n].q[0] = *(const u32x4*)(bp);
            fb[n].q[1] = *(const u32x4*)(bp + 16);
        }
        #pragma unroll
        for (int m = 0; m < 2; ++m)
            #pragma unroll
            for (int n = 0; n < 2; ++n)
                acc[m][n] = __builtin_amdgcn_wmma_f32_16x16x32_bf16(
                    false, fa[m].v, false, fb[n].v, (short)0, acc[m][n], false, false);
        __syncthreads();
    }

    // --- epilogue (C layout: vgpr r -> row r(+8), lane -> col)
    #pragma unroll
    for (int m = 0; m < 2; ++m) {
        const int row0 = tM + wm * 32 + m * 16 + khalf;
        #pragma unroll
        for (int n = 0; n < 2; ++n) {
            const int col = tN + wn * 32 + n * 16 + lr;
            const float bv = bias[col];
            #pragma unroll
            for (int r = 0; r < 8; ++r) {
                const float vacc = acc[m][n][r] + bv;
                const size_t off = (size_t)(row0 + r) * N + col;
                if (EP == 0) {
                    Cb[off] = (__bf16)vacc;
                } else if (EP == 1) {
                    Cb[off] = (__bf16)(0.5f * vacc * (1.f + erff(vacc * 0.70710678f)));
                } else {
                    Cf[off] = resid[off] + vacc;
                }
            }
        }
    }
}

// ---------------------------------------------------------------------------
// Flash attention, one wave per (b, h, 16-query tile).  Causal.
// scores: Q(16x64) @ K^T via 2 WMMA per 16-key subtile; PV: WMMA 16x16x32.
// ---------------------------------------------------------------------------
__global__ __launch_bounds__(32)
void attn_flash(const __bf16* __restrict__ Q, const __bf16* __restrict__ Kb,
                const __bf16* __restrict__ V, __bf16* __restrict__ Y)
{
    const int b = blockIdx.z, h = blockIdx.y;
    const int q0 = blockIdx.x * 16;
    const int lane = threadIdx.x & 31;
    const int lr = lane & 15;
    const int khalf = (lane < 16) ? 0 : 8;
    const size_t hoff = (size_t)h * CDH;

    __shared__ float  sS[16 * 32];
    __shared__ float  sScale[16];
    __shared__ float  sLinv[16];
    __shared__ __bf16 sP[16 * 32];
    __shared__ __bf16 sVt[64 * 32];   // [dh][key]

    // Q fragments (two 32-wide dh halves)
    FragB16 qf[2];
    {
        const __bf16* qrow = Q + ((size_t)(b * CS + q0 + lr)) * CD + hoff;
        #pragma unroll
        for (int hv = 0; hv < 2; ++hv) {
            qf[hv].q[0] = *(const u32x4*)(qrow + hv * 32 + khalf);
            qf[hv].q[1] = *(const u32x4*)(qrow + hv * 32 + khalf + 16);
        }
    }

    v8f o[4];
    #pragma unroll
    for (int nd = 0; nd < 4; ++nd)
        #pragma unroll
        for (int r = 0; r < 8; ++r) o[nd][r] = 0.f;

    float mi = -__builtin_inff();
    float li = 0.f;

    for (int j0 = 0; j0 <= q0 + 15; j0 += 32) {
        // ---- scores for 32 keys (two 16-key subtiles)
        v8f sc[2];
        #pragma unroll
        for (int kt = 0; kt < 2; ++kt) {
            #pragma unroll
            for (int r = 0; r < 8; ++r) sc[kt][r] = 0.f;
            const __bf16* krow = Kb + ((size_t)(b * CS + j0 + kt * 16 + lr)) * CD + hoff;
            #pragma unroll
            for (int hv = 0; hv < 2; ++hv) {
                FragB16 kf;
                kf.q[0] = *(const u32x4*)(krow + hv * 32 + khalf);
                kf.q[1] = *(const u32x4*)(krow + hv * 32 + khalf + 16);
                sc[kt] = __builtin_amdgcn_wmma_f32_16x16x32_bf16(false, qf[hv].v, false, kf.v,
                                                                 (short)0, sc[kt], false, false);
            }
        }
        // dump scaled scores (1/sqrt(64)=0.125) to LDS
        #pragma unroll
        for (int kt = 0; kt < 2; ++kt)
            #pragma unroll
            for (int r = 0; r < 8; ++r)
                sS[(r + khalf) * 32 + kt * 16 + lr] = sc[kt][r] * 0.125f;
        asm volatile("s_wait_dscnt 0" ::: "memory");
        __syncthreads();

        // ---- online softmax: lanes 0..15 each own one query row
        if (lane < 16) {
            const int qrow = q0 + lane;
            float tmax = mi;
            float pv[32];
            #pragma unroll
            for (int c = 0; c < 32; ++c) {
                const float s = (j0 + c <= qrow) ? sS[lane * 32 + c] : -__builtin_inff();
                pv[c] = s;
                tmax = fmaxf(tmax, s);
            }
            const float corr = __expf(mi - tmax);
            float psum = 0.f;
            #pragma unroll
            for (int c = 0; c < 32; ++c) {
                const float p = __expf(pv[c] - tmax);
                psum += p;
                sP[lane * 32 + c] = (__bf16)p;
            }
            mi = tmax;
            li = li * corr + psum;
            sScale[lane] = corr;
        }
        // ---- stage V tile transposed: [dh][key]
        {
            const __bf16* vrow = V + ((size_t)(b * CS + j0 + lane)) * CD + hoff;
            #pragma unroll
            for (int d = 0; d < 64; ++d) sVt[d * 32 + lane] = vrow[d];
        }
        __syncthreads();

        // ---- rescale running output
        #pragma unroll
        for (int r = 0; r < 8; ++r) {
            const float c0 = sScale[r + khalf];
            o[0][r] *= c0; o[1][r] *= c0; o[2][r] *= c0; o[3][r] *= c0;
        }
        // ---- P @ V
        FragB16 pf;
        pf.q[0] = *(const u32x4*)(&sP[lr * 32 + khalf]);
        pf.q[1] = *(const u32x4*)(&sP[lr * 32 + khalf + 16]);
        #pragma unroll
        for (int nd = 0; nd < 4; ++nd) {
            FragB16 vf;
            vf.q[0] = *(const u32x4*)(&sVt[(nd * 16 + lr) * 32 + khalf]);
            vf.q[1] = *(const u32x4*)(&sVt[(nd * 16 + lr) * 32 + khalf + 16]);
            o[nd] = __builtin_amdgcn_wmma_f32_16x16x32_bf16(false, pf.v, false, vf.v,
                                                            (short)0, o[nd], false, false);
        }
        __syncthreads();
    }

    if (lane < 16) sLinv[lane] = 1.f / li;
    __syncthreads();
    #pragma unroll
    for (int nd = 0; nd < 4; ++nd)
        #pragma unroll
        for (int r = 0; r < 8; ++r) {
            const int row = q0 + r + khalf;
            Y[((size_t)(b * CS) + row) * CD + hoff + nd * 16 + lr] =
                (__bf16)(o[nd][r] * sLinv[r + khalf]);
        }
}

// ---------------------------------------------------------------------------
// Head: hid = silu(xf @ Wh1 + bh1) ; out = hid @ Wh2 + bh2
// ---------------------------------------------------------------------------
__global__ __launch_bounds__(128)
void head1_kernel(const __bf16* __restrict__ hb, const float* __restrict__ Wh1,
                  const float* __restrict__ bh1, float* __restrict__ hid)
{
    const int bt = blockIdx.x;                // b*CT + s
    const int b = bt / CT, s = bt % CT;
    const size_t row = (size_t)b * CS + 1 + s;   // drop goal token
    __shared__ float xs[CD];
    for (int i = threadIdx.x; i < CD; i += 128) xs[i] = (float)hb[row * CD + i];
    __syncthreads();
    const int t = threadIdx.x;
    if (t < CHID) {
        float a = bh1[t];
        for (int k = 0; k < CD; ++k) a = fmaf(xs[k], Wh1[(size_t)k * CHID + t], a);
        hid[(size_t)bt * CHID + t] = a / (1.f + __expf(-a));   // silu
    }
}

__global__ __launch_bounds__(32)
void head2_kernel(const float* __restrict__ hid, const float* __restrict__ Wh2,
                  const float* __restrict__ bh2, float* __restrict__ out)
{
    const int bt = blockIdx.x;
    const int t = threadIdx.x;
    if (t < COUT) {
        float a = bh2[t];
        for (int k = 0; k < CHID; ++k)
            a = fmaf(hid[(size_t)bt * CHID + k], Wh2[(size_t)k * COUT + t], a);
        out[(size_t)bt * COUT + t] = a;
    }
}

// ---------------------------------------------------------------------------
// Host orchestration
// ---------------------------------------------------------------------------
extern "C" void kernel_launch(void* const* d_in, const int* in_sizes, int n_in,
                              void* d_out, int out_size, void* d_ws, size_t ws_size,
                              hipStream_t stream)
{
    (void)in_sizes; (void)n_in; (void)out_size; (void)ws_size;

    const float* states = (const float*)d_in[0];
    const float* goals  = (const float*)d_in[1];
    const float* Wtok   = (const float*)d_in[2];
    const float* btok   = (const float*)d_in[3];
    const float* pos    = (const float*)d_in[4];
    const float* ln1_g  = (const float*)d_in[5];
    const float* ln1_b  = (const float*)d_in[6];
    const float* Wq     = (const float*)d_in[7];
    const float* bq     = (const float*)d_in[8];
    const float* Wk     = (const float*)d_in[9];
    const float* bk     = (const float*)d_in[10];
    const float* Wv     = (const float*)d_in[11];
    const float* bv     = (const float*)d_in[12];
    const float* Wp     = (const float*)d_in[13];
    const float* bp     = (const float*)d_in[14];
    const float* ln2_g  = (const float*)d_in[15];
    const float* ln2_b  = (const float*)d_in[16];
    const float* Wm1    = (const float*)d_in[17];
    const float* bm1    = (const float*)d_in[18];
    const float* Wm2    = (const float*)d_in[19];
    const float* bm2    = (const float*)d_in[20];
    const float* lnf_g  = (const float*)d_in[21];
    const float* lnf_b  = (const float*)d_in[22];
    const float* Wh1    = (const float*)d_in[23];
    const float* bh1    = (const float*)d_in[24];
    const float* Wh2    = (const float*)d_in[25];
    const float* bh2    = (const float*)d_in[26];

    const size_t ROWS = (size_t)CB * CS;   // 8192
    char* w = (char*)d_ws;
    size_t off = 0;
    auto take = [&](size_t bytes) -> char* {
        char* p = w + off; off += (bytes + 255) & ~(size_t)255; return p;
    };
    float*  xf  = (float*) take(ROWS * CD * sizeof(float));
    __bf16* hb  = (__bf16*)take(ROWS * CD * sizeof(__bf16));
    __bf16* qb  = (__bf16*)take(ROWS * CD * sizeof(__bf16));
    __bf16* kb  = (__bf16*)take(ROWS * CD * sizeof(__bf16));
    __bf16* vb  = (__bf16*)take(ROWS * CD * sizeof(__bf16));
    __bf16* yb  = (__bf16*)take(ROWS * CD * sizeof(__bf16));
    float*  hid = (float*) take((size_t)CB * CT * CHID * sizeof(float));
    __bf16* mb  = qb;  // MLP hidden (8192x4096 bf16 = 64MB) aliases dead q/k/v/y

    // 1) embeddings
    embed_kernel<<<(int)ROWS, 256, 0, stream>>>(states, goals, Wtok, btok, pos, xf);

    // 2) transformer blocks
    const dim3 gD (CD   / 64, (unsigned)(ROWS / 128));   // N=1024
    const dim3 gFF(CDFF / 64, (unsigned)(ROWS / 128));   // N=4096
    const dim3 gAtt(CS / 16, CH, CB);
    for (int l = 0; l < CL; ++l) {
        const size_t dd  = (size_t)CD * CD;
        const size_t dmf = (size_t)CD * CDFF;
        ln_kernel<<<(int)ROWS, 256, 0, stream>>>(xf, ln1_g + l * CD, ln1_b + l * CD, hb);
        gemm_bf16<0><<<gD, 256, 0, stream>>>(hb, Wq + l * dd, bq + l * CD,
                                             nullptr, nullptr, qb, (int)ROWS, CD, CD);
        gemm_bf16<0><<<gD, 256, 0, stream>>>(hb, Wk + l * dd, bk + l * CD,
                                             nullptr, nullptr, kb, (int)ROWS, CD, CD);
        gemm_bf16<0><<<gD, 256, 0, stream>>>(hb, Wv + l * dd, bv + l * CD,
                                             nullptr, nullptr, vb, (int)ROWS, CD, CD);
        attn_flash<<<gAtt, 32, 0, stream>>>(qb, kb, vb, yb);
        gemm_bf16<2><<<gD, 256, 0, stream>>>(yb, Wp + l * dd, bp + l * CD,
                                             xf, xf, nullptr, (int)ROWS, CD, CD);
        ln_kernel<<<(int)ROWS, 256, 0, stream>>>(xf, ln2_g + l * CD, ln2_b + l * CD, hb);
        gemm_bf16<1><<<gFF, 256, 0, stream>>>(hb, Wm1 + l * dmf, bm1 + l * CDFF,
                                              nullptr, nullptr, mb, (int)ROWS, CDFF, CD);
        gemm_bf16<2><<<gD, 256, 0, stream>>>(mb, Wm2 + l * dmf, bm2 + l * CD,
                                             xf, xf, nullptr, (int)ROWS, CD, CDFF);
    }

    // 3) final LN + head
    ln_kernel<<<(int)ROWS, 256, 0, stream>>>(xf, lnf_g, lnf_b, hb);
    head1_kernel<<<CB * CT, 128, 0, stream>>>(hb, Wh1, bh1, hid);
    head2_kernel<<<CB * CT, 32, 0, stream>>>(hid, Wh2, bh2, (float*)d_out);
}
